// ESN_670014899083
// MI455X (gfx1250) — compile-verified, matrix operand order checked by reading, not run on hardware
//
#include <hip/hip_runtime.h>
#include <stdint.h>

// ---------------------------------------------------------------------------
// ESN: inp_proj GEMM (WMMA f16) -> persistent LDS-resident recurrence
//      (async global->LDS state broadcast) -> readout GEMM (WMMA f16)
// T=4096, INPUT_DIM=64, RES_DIM=2048, OUT_DIM=64
// ---------------------------------------------------------------------------

typedef __attribute__((ext_vector_type(16))) _Float16 v16h;
typedef __attribute__((ext_vector_type(8)))  _Float16 v8h;
typedef __attribute__((ext_vector_type(8)))  float    v8f;

#define T_STEPS 4096
#define RES     2048
#define DIN     64
#define DOUT    64

#define REC_NWG     64
#define REC_ROWS    32          // rows of W per workgroup (64*32 = 2048)
#define REC_RS      2064        // padded LDS row stride in floats (2048+16)
#define REC_THREADS 512         // == RES/4 -> one b128 async copy per thread

// ---------------------------------------------------------------------------
// CDNA5 async global->LDS copy (ASYNCcnt-tracked), 16 bytes per lane.
// lds_off: LDS byte offset (low 32 bits of the flat shared pointer).
// ---------------------------------------------------------------------------
__device__ __forceinline__ void async_copy_b128(uint32_t lds_off,
                                                const void* gaddr) {
  asm volatile("global_load_async_to_lds_b128 %0, %1, off"
               :: "v"(lds_off), "v"(gaddr) : "memory");
}
__device__ __forceinline__ void wait_asynccnt0() {
  asm volatile("s_wait_asynccnt 0x0" ::: "memory");
}

// ---------------------------------------------------------------------------
// init: zero the per-timestep barrier counters (every launch, so graph
// replays see fresh counters)
// ---------------------------------------------------------------------------
__global__ void esn_init_counters(int* __restrict__ cnt) {
  int i = blockIdx.x * 256 + threadIdx.x;
  if (i < T_STEPS) cnt[i] = 0;
}

// ---------------------------------------------------------------------------
// Convert W_out [64,2048] f32 -> f16 once (readout GEMM B operand).
// ---------------------------------------------------------------------------
__global__ void esn_cvt_wout(const float* __restrict__ Wout,
                             _Float16* __restrict__ Wout16) {
  int i = blockIdx.x * 256 + threadIdx.x;   // 131072 elements
  Wout16[i] = (_Float16)Wout[i];
}

// ---------------------------------------------------------------------------
// inp_proj = u @ W_in^T  : [4096,64] x [64,2048] -> P [4096,2048] (f32)
// One 16x16 tile per wave, K=64 -> two v_wmma_f32_16x16x32_f16.
// Per-lane A/B element s holds k = 8*half + s (+8 for s >= 8): packed float4 loads + cvt.
// ---------------------------------------------------------------------------
__global__ void esn_inproj_wmma(const float* __restrict__ u,
                                const float* __restrict__ Win,
                                float* __restrict__ P) {
  const int lane   = threadIdx.x & 31;
  const int wave   = threadIdx.x >> 5;
  const int tile   = blockIdx.x * 8 + wave;   // 256*128 tiles / 8 waves
  const int tm     = tile >> 7;               // 0..255
  const int tn     = tile & 127;              // 0..127
  const int half   = lane >> 4;
  const int lane16 = lane & 15;
  const int m = tm * 16 + lane16;
  const int n = tn * 16 + lane16;

  v8f c = {};
  for (int kb = 0; kb < 2; ++kb) {
    const float* ua = u   + m * DIN + kb * 32 + half * 8;
    const float* wb = Win + n * DIN + kb * 32 + half * 8;
    const float4 a0 = *(const float4*)(ua);
    const float4 a1 = *(const float4*)(ua + 4);
    const float4 a2 = *(const float4*)(ua + 16);
    const float4 a3 = *(const float4*)(ua + 20);
    const float4 b0 = *(const float4*)(wb);
    const float4 b1 = *(const float4*)(wb + 4);
    const float4 b2 = *(const float4*)(wb + 16);
    const float4 b3 = *(const float4*)(wb + 20);
    v16h a, b;
    a[0]=(_Float16)a0.x; a[1]=(_Float16)a0.y; a[2]=(_Float16)a0.z; a[3]=(_Float16)a0.w;
    a[4]=(_Float16)a1.x; a[5]=(_Float16)a1.y; a[6]=(_Float16)a1.z; a[7]=(_Float16)a1.w;
    a[8]=(_Float16)a2.x; a[9]=(_Float16)a2.y; a[10]=(_Float16)a2.z; a[11]=(_Float16)a2.w;
    a[12]=(_Float16)a3.x; a[13]=(_Float16)a3.y; a[14]=(_Float16)a3.z; a[15]=(_Float16)a3.w;
    b[0]=(_Float16)b0.x; b[1]=(_Float16)b0.y; b[2]=(_Float16)b0.z; b[3]=(_Float16)b0.w;
    b[4]=(_Float16)b1.x; b[5]=(_Float16)b1.y; b[6]=(_Float16)b1.z; b[7]=(_Float16)b1.w;
    b[8]=(_Float16)b2.x; b[9]=(_Float16)b2.y; b[10]=(_Float16)b2.z; b[11]=(_Float16)b2.w;
    b[12]=(_Float16)b3.x; b[13]=(_Float16)b3.y; b[14]=(_Float16)b3.z; b[15]=(_Float16)b3.w;
    c = __builtin_amdgcn_wmma_f32_16x16x32_f16(false, a, false, b,
                                               (short)0, c, false, false);
  }
#pragma unroll
  for (int r = 0; r < 8; ++r) {
    const int mr = tm * 16 + half * 8 + r;    // C/D layout: M = (lane/16)*8 + r
    P[mr * RES + tn * 16 + lane16] = c[r];
  }
}

// ---------------------------------------------------------------------------
// Persistent recurrence: s_t = tanh(P[t] + W s_{t-1})
// 64 WGs, each owns 32 rows of W resident in LDS for all 4096 steps.
// State broadcast uses async global->LDS b128 copies (ASYNCcnt).
// ---------------------------------------------------------------------------
__global__ void esn_recurrence(const float* __restrict__ W,
                               const float* __restrict__ P,
                               float* __restrict__ gs,         // 2*RES f32
                               _Float16* __restrict__ S16,     // [T,RES] f16
                               int* __restrict__ cnt) {        // [T]
  extern __shared__ float lds[];
  float* Wl = lds;                       // REC_ROWS * REC_RS
  float* sl = lds + REC_ROWS * REC_RS;   // RES floats (prev state)

  const int tid     = threadIdx.x;
  const int rowbase = blockIdx.x * REC_ROWS;

  // Preload this WG's 32 rows of W into LDS (padded row stride).
  for (int i = tid; i < REC_ROWS * RES / 4; i += REC_THREADS) {
    const int e = i * 4;
    const int r = e >> 11;        // /2048
    const int col = e & (RES - 1);
    const float4 w = *(const float4*)(W + (size_t)(rowbase + r) * RES + col);
    *(float4*)(Wl + r * REC_RS + col) = w;
  }
  // s_{-1} = 0 -> step 0 reduces to tanh(P[0]).
  for (int i = tid; i < RES; i += REC_THREADS) sl[i] = 0.0f;
  __syncthreads();

  const int tx  = tid & 15;       // 16 lanes per row
  const int row = tid >> 4;       // 0..31
  const float4* wrow = (const float4*)(Wl + row * REC_RS);
  const float4* s4   = (const float4*)sl;
  // LDS byte offset of this thread's 16-byte slot in sl (flat ptr low bits).
  const uint32_t sl_off = (uint32_t)(uintptr_t)(sl + tid * 4);

  for (int t = 0; t < T_STEPS; ++t) {
    if (t > 0) {
      if (tid == 0) {
        while (__hip_atomic_load(&cnt[t - 1], __ATOMIC_ACQUIRE,
                                 __HIP_MEMORY_SCOPE_AGENT) < REC_NWG) {
          __builtin_amdgcn_s_sleep(2);
        }
      }
      __syncthreads();
      // Broadcast s_{t-1} (8 KB) L2 -> LDS: one async b128 copy per thread.
      async_copy_b128(sl_off, gs + ((t - 1) & 1) * RES + tid * 4);
      wait_asynccnt0();
      __syncthreads();
    }

    // 32 rows x 2048 dot products. k = 4*tx + 64*j covers all 64 banks
    // across the 16 lanes of a row (b128 LDS loads).
    float acc = 0.0f;
#pragma unroll 8
    for (int j = 0; j < 32; ++j) {
      const float4 w = wrow[tx + 16 * j];
      const float4 s = s4[tx + 16 * j];
      acc = fmaf(w.x, s.x, acc);
      acc = fmaf(w.y, s.y, acc);
      acc = fmaf(w.z, s.z, acc);
      acc = fmaf(w.w, s.w, acc);
    }
    // Reduce the 16 partials of each row (wave32: two rows per wave).
#pragma unroll
    for (int off = 8; off > 0; off >>= 1) acc += __shfl_xor(acc, off, 16);

    if (tx == 0) {
      const int rg = rowbase + row;
      const float v = tanhf(P[(size_t)t * RES + rg] + acc);
      gs[(t & 1) * RES + rg] = v;                 // f32 for next step
      S16[(size_t)t * RES + rg] = (_Float16)v;    // f16 for readout GEMM
      __threadfence();                            // release our stores to L2
      // Warm L2/L0 for next step's bias element during the upcoming spin.
      const int tn_ = (t + 1 < T_STEPS) ? (t + 1) : t;
      __builtin_prefetch(P + (size_t)tn_ * RES + rg, 0, 1);
    }
    __syncthreads();
    if (tid == 0) {
      __hip_atomic_fetch_add(&cnt[t], 1, __ATOMIC_RELEASE,
                             __HIP_MEMORY_SCOPE_AGENT);
    }
  }
}

// ---------------------------------------------------------------------------
// out = states @ W_out^T + b : [4096,2048](f16) x [2048,64](f16) -> f32
// One 16x16 tile per wave, 64 x v_wmma_f32_16x16x32_f16 along K.
// Packed v8h (16B) loads for both operands.
// ---------------------------------------------------------------------------
__global__ void esn_readout_wmma(const _Float16* __restrict__ S16,
                                 const _Float16* __restrict__ Wout16,
                                 const float* __restrict__ bout,
                                 float* __restrict__ out) {
  const int lane   = threadIdx.x & 31;
  const int wave   = threadIdx.x >> 5;
  const int tile   = blockIdx.x * 8 + wave;   // 256*4 tiles / 8 waves
  const int tm     = tile >> 2;               // 0..255
  const int tn     = tile & 3;                // 0..3
  const int half   = lane >> 4;
  const int lane16 = lane & 15;
  const _Float16* arow = S16    + (size_t)(tm * 16 + lane16) * RES + half * 8;
  const _Float16* brow = Wout16 + (size_t)(tn * 16 + lane16) * RES + half * 8;

  v8f c = {};
  for (int kb = 0; kb < RES / 32; ++kb) {
    const v8h alo = *(const v8h*)(arow + kb * 32);
    const v8h ahi = *(const v8h*)(arow + kb * 32 + 16);
    const v8h blo = *(const v8h*)(brow + kb * 32);
    const v8h bhi = *(const v8h*)(brow + kb * 32 + 16);
    v16h a, b;
#pragma unroll
    for (int i = 0; i < 8; ++i) {
      a[i] = alo[i]; a[i + 8] = ahi[i];
      b[i] = blo[i]; b[i + 8] = bhi[i];
    }
    c = __builtin_amdgcn_wmma_f32_16x16x32_f16(false, a, false, b,
                                               (short)0, c, false, false);
  }
#pragma unroll
  for (int r = 0; r < 8; ++r) {
    const int mr = tm * 16 + half * 8 + r;
    const int nc = tn * 16 + lane16;
    out[mr * DOUT + nc] = c[r] + bout[nc];
  }
}

// ---------------------------------------------------------------------------
extern "C" void kernel_launch(void* const* d_in, const int* in_sizes, int n_in,
                              void* d_out, int out_size, void* d_ws, size_t ws_size,
                              hipStream_t stream) {
  const float* u    = (const float*)d_in[0];   // [4096,64]
  const float* Win  = (const float*)d_in[1];   // [2048,64]
  const float* W    = (const float*)d_in[2];   // [2048,2048]
  const float* Wout = (const float*)d_in[3];   // [64,2048]
  const float* bout = (const float*)d_in[4];   // [64]
  float* out = (float*)d_out;                  // [4096,64]

  // Workspace layout
  char* ws = (char*)d_ws;
  float*    P      = (float*)ws;                                   // 32 MB
  _Float16* S16    = (_Float16*)(ws + (size_t)T_STEPS * RES * 4);  // 16 MB
  char*     after  = ws + (size_t)T_STEPS * RES * 4 + (size_t)T_STEPS * RES * 2;
  float*    gs     = (float*)after;                                // 16 KB
  int*      cnt    = (int*)(after + 2 * RES * sizeof(float));      // 16 KB
  _Float16* Wout16 = (_Float16*)(after + 2 * RES * sizeof(float)
                                       + T_STEPS * sizeof(int));   // 256 KB

  esn_init_counters<<<(T_STEPS + 255) / 256, 256, 0, stream>>>(cnt);
  esn_cvt_wout<<<(DOUT * RES) / 256, 256, 0, stream>>>(Wout, Wout16);

  // inp_proj: 256*128 tiles, 8 waves/block -> 4096 blocks
  esn_inproj_wmma<<<4096, 256, 0, stream>>>(u, Win, P);

  // persistent recurrence: 64 WGs x 512 threads, 264KB W + 8KB s in LDS
  const size_t ldsBytes = (size_t)(REC_ROWS * REC_RS + RES) * sizeof(float);
  hipFuncSetAttribute(reinterpret_cast<const void*>(esn_recurrence),
                      hipFuncAttributeMaxDynamicSharedMemorySize,
                      (int)ldsBytes);
  esn_recurrence<<<REC_NWG, REC_THREADS, ldsBytes, stream>>>(W, P, gs, S16, cnt);

  // readout: 256*4 tiles, 8 waves/block -> 128 blocks
  esn_readout_wmma<<<128, 256, 0, stream>>>(S16, Wout16, bout, out);
}
